// SparseDownBlock3d_12970801233922
// MI455X (gfx1250) — matrix operand order repaired and checked by reading, not compiled, not run on hardware
//
#include <hip/hip_runtime.h>

// SparseDownBlock3d for MI455X (gfx1250, wave32).
// Convs run as implicit GEMM via v_wmma_f32_16x16x32_bf16 (f32 accumulate).
// Tile staging uses gfx1250 async global->LDS DMA (ASYNCcnt) where it is a
// pure copy; GN+SiLU-fused staging stays on the VALU path.
// B=4, D=64 -> Dd=32, C=64, Co=128, G=32 groups.

typedef __attribute__((ext_vector_type(16))) __bf16 v16bf;
typedef __attribute__((ext_vector_type(8)))  float  v8f;

#define EPSV 1e-5f

__device__ __forceinline__ unsigned short f2bf(float f) {
  unsigned u = __builtin_bit_cast(unsigned, f);
  u += 0x7FFFu + ((u >> 16) & 1u);          // round-to-nearest-even
  return (unsigned short)(u >> 16);
}
__device__ __forceinline__ float bf2f(unsigned short h) {
  return __builtin_bit_cast(float, ((unsigned)h) << 16);
}
__device__ __forceinline__ unsigned packbf(float a, float b) {
  return (unsigned)f2bf(a) | ((unsigned)f2bf(b) << 16);
}
__device__ __forceinline__ float siluf(float v) { return v / (1.f + __expf(-v)); }

// per-lane async DMA: 16 bytes global -> LDS, tracked by ASYNCcnt
__device__ __forceinline__ void async_copy_b128(unsigned lds_addr,
                                                unsigned long long gaddr) {
  asm volatile("global_load_async_to_lds_b128 %0, %1, off"
               :: "v"(lds_addr), "v"(gaddr) : "memory");
}
__device__ __forceinline__ void wait_async0() {
  asm volatile("s_wait_asynccnt 0" ::: "memory");
}

union FragBF { uint4 u[2]; v16bf v; };
union FragF  { float f[8];  v8f   v; };

// ---------------- GN1 stats: masked sum / sumsq per (b, g), count per b -------
__global__ __launch_bounds__(256) void k_gn1_stats(
    const float* __restrict__ x, const int* __restrict__ mask,
    float* gsum, float* gsumsq, float* gcnt) {
  __shared__ float lsum[32], lss[32];
  __shared__ float lcnt;
  int tid = threadIdx.x;
  if (tid < 32) { lsum[tid] = 0.f; lss[tid] = 0.f; }
  if (tid == 0) lcnt = 0.f;
  __syncthreads();
  int b = blockIdx.x >> 9;            // 512 blocks per batch
  int chunk = blockIdx.x & 511;       // 512 voxels per block
  int g  = tid & 31;                  // group == channel pair (Cg = 2)
  int vs = tid >> 5;                  // 0..7 voxels in flight
  const float2* xp = (const float2*)x;
  float s = 0.f, ss = 0.f, cnt = 0.f;
  int vbase = chunk * 512;
  for (int it = 0; it < 64; ++it) {
    int v = vbase + it * 8 + vs;
    float mm = (float)mask[b * 262144 + v];
    float2 xv = xp[(size_t)(b * 262144 + v) * 32 + g];
    float a = xv.x * mm, c = xv.y * mm;
    s += a + c; ss += a * a + c * c;
    if (g == 0) cnt += mm;
  }
  atomicAdd(&lsum[g], s);
  atomicAdd(&lss[g], ss);
  if (g == 0) atomicAdd(&lcnt, cnt);
  __syncthreads();
  if (tid < 32) {
    atomicAdd(&gsum[b * 32 + tid], lsum[tid]);
    atomicAdd(&gsumsq[b * 32 + tid], lss[tid]);
  }
  if (tid == 0) atomicAdd(&gcnt[b], lcnt);
}

// ------------- pack weights into WMMA B-fragment layout (bf16) ---------------
// dst[(((tap*ksteps+ks)*ntiles+nt)*32+lane)*8+dword] ; lane={half,n}, K=half*16+2*dword
__global__ void k_pack(const float* __restrict__ W, unsigned* __restrict__ dst,
                       int taps, int cin, int cout) {
  int idx = blockIdx.x * 256 + threadIdx.x;
  int ksteps = cin >> 5, ntiles = cout >> 4;
  int total = taps * ksteps * ntiles * 256;
  if (idx >= total) return;
  int dword = idx & 7;
  int lane  = (idx >> 3) & 31;
  int blk   = idx >> 8;
  int nt  = blk % ntiles;
  int ks  = (blk / ntiles) % ksteps;
  int tap = blk / (ntiles * ksteps);
  int half = lane >> 4, n = lane & 15;
  int c = ks * 32 + half * 16 + dword * 2;
  float v0 = W[(size_t)(tap * cin + c)     * cout + nt * 16 + n];
  float v1 = W[(size_t)(tap * cin + c + 1) * cout + nt * 16 + n];
  dst[idx] = packbf(v0, v1);
}

// ------- fused GN1 normalize + SiLU + 2x downsample -> h_down, xd (bf16) -----
__global__ __launch_bounds__(256) void k_down(
    const float* __restrict__ x, const int* __restrict__ mask,
    const float* __restrict__ gw, const float* __restrict__ gb,
    const float* gsum, const float* gsumsq, const float* gcnt,
    unsigned* __restrict__ hdown, unsigned* __restrict__ xd,
    float* __restrict__ dm, float* gcnt2) {
  __shared__ float lcnt2;
  int tid = threadIdx.x;
  if (tid == 0) lcnt2 = 0.f;
  __syncthreads();
  int vo = blockIdx.x * 8 + (tid >> 5);   // output voxel (8 per block)
  int b  = vo >> 15;
  int r  = vo & 32767;
  int zo = r >> 10, yo = (r >> 5) & 31, xo = r & 31;
  int g  = tid & 31;
  int c0 = g * 2;
  float cntv = fmaxf(gcnt[b] * 2.f, 1.f);
  float mean = gsum[b * 32 + g] / cntv;
  float var  = gsumsq[b * 32 + g] / cntv - mean * mean;
  float inv  = rsqrtf(var + EPSV);
  float g0 = gw[c0], g1 = gw[c0 + 1], be0 = gb[c0], be1 = gb[c0 + 1];
  const float2* xp = (const float2*)x;
  float hs0 = 0, hs1 = 0, xs0 = 0, xs1 = 0, cc = 0;
  for (int ch = 0; ch < 8; ++ch) {
    int zi = zo * 2 + (ch >> 2), yi = yo * 2 + ((ch >> 1) & 1), xi = xo * 2 + (ch & 1);
    int v = ((zi << 6) + yi) * 64 + xi;
    if (mask[b * 262144 + v] != 0) {
      float2 xv = xp[(size_t)(b * 262144 + v) * 32 + g];
      hs0 += siluf((xv.x - mean) * inv * g0 + be0);
      hs1 += siluf((xv.y - mean) * inv * g1 + be1);
      xs0 += xv.x; xs1 += xv.y;
      cc += 1.f;
    }
  }
  float dnm = fmaxf(cc, 1.f);
  hdown[(size_t)vo * 32 + g] = packbf(hs0 / dnm, hs1 / dnm);
  xd[(size_t)vo * 32 + g]    = packbf(xs0 / dnm, xs1 / dnm);
  if (g == 0) {
    float dv = (cc > 0.f) ? 1.f : 0.f;
    dm[vo] = dv;
    atomicAdd(&lcnt2, dv);
  }
  __syncthreads();
  if (tid == 0) atomicAdd(&gcnt2[b], lcnt2);
}

// ---------- conv1 64->128 implicit GEMM (bf16 WMMA) + fused GN2 stats --------
__global__ __launch_bounds__(256) void k_conv1(
    const uint4* __restrict__ hdown, const uint4* __restrict__ pw,
    const float* __restrict__ bias, const float* __restrict__ dm,
    unsigned short* __restrict__ y1, float* gsum2, float* gss2) {
  __shared__ uint4 tile[162 * 8];            // 3x3x18 voxels x 64ch bf16 (20.7 KB)
  __shared__ float lsum[32], lss[32];
  int tid = threadIdx.x;
  if (tid < 32) { lsum[tid] = 0.f; lss[tid] = 0.f; }
  int b   = blockIdx.x >> 11;
  int rem = blockIdx.x & 2047;
  int zo = rem >> 6, yo = (rem >> 1) & 31, w0 = (rem & 1) * 16;
  // stage halo tile: async DMA for in-bounds lanes, zero-store for halo OOB
  unsigned ldsbase = (unsigned)(size_t)(&tile[0]);
  unsigned long long gbase = (unsigned long long)(size_t)hdown;
  for (int idx = tid; idx < 162 * 8; idx += 256) {
    int t = idx >> 3, q = idx & 7;
    int dz = t / 54, rr = t % 54, dy = rr / 18, wx = rr % 18;
    int zi = zo + dz - 1, yi = yo + dy - 1, wi = w0 + wx - 1;
    if (zi >= 0 && zi < 32 && yi >= 0 && yi < 32 && wi >= 0 && wi < 32) {
      unsigned long long ga = gbase +
          (unsigned long long)(b * 32768 + ((zi * 32 + yi) * 32 + wi)) * 128ull +
          (unsigned long long)(q * 16);
      async_copy_b128(ldsbase + (unsigned)idx * 16u, ga);
    } else {
      tile[idx] = make_uint4(0, 0, 0, 0);
    }
  }
  wait_async0();
  __syncthreads();
  int lane = tid & 31, nt = tid >> 5;        // 8 waves -> 8 oc tiles of 16
  int half = lane >> 4, m = lane & 15, n = lane & 15;
  FragF acc = {};
  const char* tbase = (const char*)tile;
  for (int tap = 0; tap < 27; ++tap) {
    int dz = tap / 9, dy = (tap / 3) % 3, dx = tap % 3;
    int vb = (((dz * 3 + dy) * 18) + dx + m) * 128;
    for (int ks = 0; ks < 2; ++ks) {
      FragBF a;
      const uint4* ap = (const uint4*)(tbase + vb + ks * 64 + half * 32);
      a.u[0] = ap[0]; a.u[1] = ap[1];        // channels +0..7 and +8..15
      FragBF bm;
      const uint4* bp = pw + (size_t)(((tap * 2 + ks) * 8 + nt) * 64 + lane * 2);
      bm.u[0] = bp[0]; bm.u[1] = bp[1];
      acc.v = __builtin_amdgcn_wmma_f32_16x16x32_bf16(
          false, a.v, false, bm.v, (short)0, acc.v, false, false);
    }
  }
  int oc  = nt * 16 + n;
  float bsv = bias[oc];
  int vrow = b * 32768 + ((zo * 32 + yo) * 32) + w0;
  int g2 = oc >> 2;                          // Cg = 4 for GN2
  float s = 0.f, ss = 0.f;
  for (int rr = 0; rr < 8; ++rr) {
    int mm = rr + half * 8;
    float dmv = dm[vrow + mm];
    float val = (acc.f[rr] + bsv) * dmv;
    y1[(size_t)(vrow + mm) * 128 + oc] = f2bf(val);
    s += val; ss += val * val;
  }
  atomicAdd(&lsum[g2], s);
  atomicAdd(&lss[g2], ss);
  __syncthreads();
  if (tid < 32) {
    atomicAdd(&gsum2[b * 32 + tid], lsum[tid]);
    atomicAdd(&gss2[b * 32 + tid], lss[tid]);
  }
}

// -- conv2 128->128 (GN2+SiLU fused into staging) + fused 1x1 skip GEMM -------
__global__ __launch_bounds__(256) void k_conv2(
    const unsigned* __restrict__ y1, const unsigned* __restrict__ xdv,
    const uint4* __restrict__ pw2, const uint4* __restrict__ pws,
    const float* __restrict__ b2, const float* __restrict__ bsk,
    const float* __restrict__ g2w, const float* __restrict__ g2b,
    const float* gsum2, const float* gss2, const float* gcnt2,
    const float* __restrict__ dm, float* __restrict__ out) {
  __shared__ __attribute__((aligned(16))) unsigned tile[162 * 64 + 16 * 32];
  __shared__ float smean[32], sinv[32], sgam[128], sbet[128];
  int tid = threadIdx.x;
  int b   = blockIdx.x >> 11;
  int rem = blockIdx.x & 2047;
  int zo = rem >> 6, yo = (rem >> 1) & 31, w0 = (rem & 1) * 16;
  // kick off the 2KB xd skip-tile DMA first: it overlaps the VALU-heavy
  // GN2+SiLU staging below (ASYNCcnt-tracked, waited before the barrier)
  if (tid < 128) {
    unsigned lds_skip = (unsigned)(size_t)(&tile[162 * 64]) + (unsigned)tid * 16u;
    int t = tid >> 3, q = tid & 7;
    unsigned long long ga = (unsigned long long)(size_t)xdv +
        (unsigned long long)(b * 32768 + ((zo * 32 + yo) * 32) + w0 + t) * 128ull +
        (unsigned long long)(q * 16);
    async_copy_b128(lds_skip, ga);
  }
  if (tid < 32) {
    float cntv = fmaxf(gcnt2[b] * 4.f, 1.f);
    float mean = gsum2[b * 32 + tid] / cntv;
    float var  = gss2[b * 32 + tid] / cntv - mean * mean;
    smean[tid] = mean;
    sinv[tid]  = rsqrtf(var + EPSV);
  }
  if (tid < 128) { sgam[tid] = g2w[tid]; sbet[tid] = g2b[tid]; }
  __syncthreads();
  for (int idx = tid; idx < 162 * 64; idx += 256) {   // GN2 + SiLU while staging
    int t = idx >> 6, dwi = idx & 63;
    int dz = t / 54, rr = t % 54, dy = rr / 18, wx = rr % 18;
    int zi = zo + dz - 1, yi = yo + dy - 1, wi = w0 + wx - 1;
    unsigned val = 0u;
    if (zi >= 0 && zi < 32 && yi >= 0 && yi < 32 && wi >= 0 && wi < 32) {
      unsigned raw = y1[(size_t)(b * 32768 + ((zi * 32 + yi) * 32 + wi)) * 64 + dwi];
      int c0 = dwi * 2, g = c0 >> 2;
      float a0 = bf2f((unsigned short)(raw & 0xFFFFu));
      float a1 = bf2f((unsigned short)(raw >> 16));
      a0 = siluf((a0 - smean[g]) * sinv[g] * sgam[c0]     + sbet[c0]);
      a1 = siluf((a1 - smean[g]) * sinv[g] * sgam[c0 + 1] + sbet[c0 + 1]);
      val = packbf(a0, a1);
    }
    tile[idx] = val;
  }
  wait_async0();
  __syncthreads();
  int lane = tid & 31, nt = tid >> 5;
  int half = lane >> 4, m = lane & 15, n = lane & 15;
  FragF acc = {};
  const char* tbase = (const char*)tile;
  for (int tap = 0; tap < 27; ++tap) {
    int dz = tap / 9, dy = (tap / 3) % 3, dx = tap % 3;
    int vb = (((dz * 3 + dy) * 18) + dx + m) * 256;
    for (int ks = 0; ks < 4; ++ks) {
      FragBF a;
      const uint4* ap = (const uint4*)(tbase + vb + ks * 64 + half * 32);
      a.u[0] = ap[0]; a.u[1] = ap[1];
      FragBF bm;
      const uint4* bp = pw2 + (size_t)(((tap * 4 + ks) * 8 + nt) * 64 + lane * 2);
      bm.u[0] = bp[0]; bm.u[1] = bp[1];
      acc.v = __builtin_amdgcn_wmma_f32_16x16x32_bf16(
          false, a.v, false, bm.v, (short)0, acc.v, false, false);
    }
  }
  const char* sbase = (const char*)(tile + 162 * 64);   // 1x1 skip: xd @ wskip
  for (int ks = 0; ks < 2; ++ks) {
    FragBF a;
    const uint4* ap = (const uint4*)(sbase + m * 128 + ks * 64 + half * 32);
    a.u[0] = ap[0]; a.u[1] = ap[1];
    FragBF bm;
    const uint4* bp = pws + (size_t)((ks * 8 + nt) * 64 + lane * 2);
    bm.u[0] = bp[0]; bm.u[1] = bp[1];
    acc.v = __builtin_amdgcn_wmma_f32_16x16x32_bf16(
        false, a.v, false, bm.v, (short)0, acc.v, false, false);
  }
  int oc = nt * 16 + n;
  float bsv = b2[oc] + bsk[oc];
  int vrow = b * 32768 + ((zo * 32 + yo) * 32) + w0;
  for (int r = 0; r < 8; ++r) {
    int mm = r + half * 8;
    out[(size_t)(vrow + mm) * 128 + oc] = (acc.f[r] + bsv) * dm[vrow + mm];
  }
}

// ------------------------------- launcher ------------------------------------
extern "C" void kernel_launch(void* const* d_in, const int* in_sizes, int n_in,
                              void* d_out, int out_size, void* d_ws, size_t ws_size,
                              hipStream_t stream) {
  const float* x    = (const float*)d_in[0];
  const int*   mask = (const int*)d_in[1];
  const float* gn1w = (const float*)d_in[2];
  const float* gn1b = (const float*)d_in[3];
  const float* w1   = (const float*)d_in[4];
  const float* b1   = (const float*)d_in[5];
  const float* gn2w = (const float*)d_in[6];
  const float* gn2b = (const float*)d_in[7];
  const float* w2   = (const float*)d_in[8];
  const float* b2   = (const float*)d_in[9];
  const float* wsk  = (const float*)d_in[10];
  const float* bsk  = (const float*)d_in[11];

  char* ws = (char*)d_ws;
  const size_t OFF_HD  = 2560;                       // stats region = [0,2560)
  const size_t OFF_XD  = OFF_HD + 16777216;          // h_down bf16
  const size_t OFF_DM  = OFF_XD + 16777216;          // xd bf16
  const size_t OFF_Y1  = OFF_DM + 524288;            // dm f32
  const size_t OFF_PW1 = OFF_Y1 + 33554432;          // y1 bf16
  const size_t OFF_PW2 = OFF_PW1 + 442368;
  const size_t OFF_PWS = OFF_PW2 + 884736;

  float* gn1_sum = (float*)(ws + 0);
  float* gn1_ss  = (float*)(ws + 512);
  float* cnt1    = (float*)(ws + 1024);
  float* gn2_sum = (float*)(ws + 1280);
  float* gn2_ss  = (float*)(ws + 1792);
  float* cnt2    = (float*)(ws + 2304);
  unsigned* hdown = (unsigned*)(ws + OFF_HD);
  unsigned* xd    = (unsigned*)(ws + OFF_XD);
  float*    dm    = (float*)(ws + OFF_DM);
  unsigned short* y1 = (unsigned short*)(ws + OFF_Y1);
  unsigned* pw1   = (unsigned*)(ws + OFF_PW1);
  unsigned* pw2   = (unsigned*)(ws + OFF_PW2);
  unsigned* pwsk  = (unsigned*)(ws + OFF_PWS);

  hipMemsetAsync(d_ws, 0, 2560, stream);             // zero stats accumulators

  k_gn1_stats<<<2048, 256, 0, stream>>>(x, mask, gn1_sum, gn1_ss, cnt1);
  k_pack<<<(27 * 2 * 8 * 256) / 256, 256, 0, stream>>>(w1, pw1, 27, 64, 128);
  k_pack<<<(27 * 4 * 8 * 256) / 256, 256, 0, stream>>>(w2, pw2, 27, 128, 128);
  k_pack<<<(1 * 2 * 8 * 256) / 256, 256, 0, stream>>>(wsk, pwsk, 1, 64, 128);
  k_down<<<16384, 256, 0, stream>>>(x, mask, gn1w, gn1b, gn1_sum, gn1_ss, cnt1,
                                    hdown, xd, dm, cnt2);
  k_conv1<<<8192, 256, 0, stream>>>((const uint4*)hdown, (const uint4*)pw1, b1,
                                    dm, y1, gn2_sum, gn2_ss);
  k_conv2<<<8192, 256, 0, stream>>>((const unsigned*)y1, xd, (const uint4*)pw2,
                                    (const uint4*)pwsk, b2, bsk, gn2w, gn2b,
                                    gn2_sum, gn2_ss, cnt2, dm, (float*)d_out);
}
